// InteractionKernel_60627758350471
// MI455X (gfx1250) — matrix operand order repaired and checked by reading, not compiled, not run on hardware
//
#include <hip/hip_runtime.h>

// Problem constants
#define BN 4
#define MN 128
#define EN 768
#define DN 768                 // E == R == 768
#define SN (MN * (MN + 1) / 2) // 8256
#define NW 589824              // 768*768 weight elements

typedef __attribute__((ext_vector_type(16))) __bf16 v16bf;
typedef __attribute__((ext_vector_type(8)))  float  v8f;

__device__ __forceinline__ unsigned short f2bf(float f) {
  union { float f; unsigned int u; } c; c.f = f;
  unsigned int u = c.u;
  u += 0x7FFFu + ((u >> 16) & 1u);   // round-to-nearest-even
  return (unsigned short)(u >> 16);
}

// (m, e) -> fragment-order LDS index for the 16x32 bf16 A layout:
// lane = m + 16*[(e&16) != 0], element p = e&15
__device__ __forceinline__ int a_lds_index(int m, int e) {
  const int kk = e >> 5;
  const int ko = e & 31;
  const int lane = m + ((ko & 16) ? 16 : 0);
  const int p = ko & 15;
  return kk * 512 + lane * 16 + p;
}

// ---------------------------------------------------------------------------
// 0) One-time f32 -> bf16 weight conversion (2 elements per thread, packed)
// ---------------------------------------------------------------------------
__global__ void __launch_bounds__(256)
k_cvt_bf16(const float* __restrict__ src, unsigned int* __restrict__ dst)
{
  const int i = blockIdx.x * 256 + threadIdx.x;       // over NW/2 pairs
  const float lo = src[2 * i];
  const float hi = src[2 * i + 1];
  dst[i] = (unsigned int)f2bf(lo) | ((unsigned int)f2bf(hi) << 16);
}

// ---------------------------------------------------------------------------
// 1) erowT[b,p,e] = sum_q ent_hs[b, mirror[p,q], e] * alpha[e,q]
//    ecolT[b,p,e] = sum_q ent_hs[b, mirror[p,q], e] * beta [e,q]
// ---------------------------------------------------------------------------
__global__ void __launch_bounds__(256)
k_bilin_ent(const float* __restrict__ ent_hs, const float* __restrict__ alpha,
            const float* __restrict__ beta, const int* __restrict__ mirror,
            float* __restrict__ erowT, float* __restrict__ ecolT)
{
  __shared__ int s_mid[MN];
  const int bp = blockIdx.x;        // b*M + p
  const int b  = bp >> 7;
  const int p  = bp & (MN - 1);
  if (threadIdx.x < MN) s_mid[threadIdx.x] = mirror[p * MN + threadIdx.x];
  __syncthreads();

  const float* hsb = ent_hs + (size_t)b * SN * EN;
  for (int ec = 0; ec < EN / 256; ++ec) {
    const int e = ec * 256 + threadIdx.x;
    float ar = 0.f, ac = 0.f;
    for (int q = 0; q < MN; ++q) {
      const float h = hsb[(size_t)s_mid[q] * EN + e];
      ar += h * alpha[(size_t)e * MN + q];
      ac += h * beta[(size_t)e * MN + q];
    }
    erowT[(size_t)bp * EN + e] = ar;
    ecolT[(size_t)bp * EN + e] = ac;
  }
}

// ---------------------------------------------------------------------------
// 2) Per-row mean and 1/((var+eps)^2) over width 768 (faithful to reference)
// ---------------------------------------------------------------------------
__global__ void __launch_bounds__(256)
k_row_stats(const float* __restrict__ x, float* __restrict__ mean_o,
            float* __restrict__ rstd_o)
{
  __shared__ float sh[256], sh2[256];
  const long row = blockIdx.x;
  const float* p = x + row * DN;
  float s = 0.f, s2 = 0.f;
  for (int e = threadIdx.x; e < DN; e += 256) {
    const float v = p[e];
    s += v; s2 += v * v;
  }
  sh[threadIdx.x] = s; sh2[threadIdx.x] = s2;
  __syncthreads();
  for (int off = 128; off > 0; off >>= 1) {
    if (threadIdx.x < off) {
      sh[threadIdx.x]  += sh[threadIdx.x + off];
      sh2[threadIdx.x] += sh2[threadIdx.x + off];
    }
    __syncthreads();
  }
  if (threadIdx.x == 0) {
    const float mean = sh[0] * (1.f / DN);
    const float var  = sh2[0] * (1.f / DN) - mean * mean;
    float stdv = var + 1e-12f; stdv = stdv * stdv;   // (var+eps)^2 per reference
    mean_o[row] = mean;
    rstd_o[row] = 1.f / stdv;
  }
}

// ---------------------------------------------------------------------------
// Shared WMMA core, 512 threads = 16 waves x 3 column tiles.
// Two-stage software pipeline over K: fragments for kk+1 are loaded while
// the WMMAs for kk execute (loop-carried dataflow forces outstanding loads,
// so waits become partial counts instead of s_wait_loadcnt 0).
// ---------------------------------------------------------------------------
__device__ __forceinline__ void cln_wave_core(
    const unsigned short* A_lds, const float* s_mean, const float* s_rstd,
    const unsigned short* __restrict__ Wb, const unsigned short* __restrict__ Wg,
    const float* __restrict__ wbeta, const float* __restrict__ wgamma,
    const float* __restrict__ xin, float* __restrict__ xout, long r0)
{
  const int tid   = threadIdx.x;
  const int lane  = tid & 31;
  const int wv    = tid >> 5;        // 0..15
  const int lm    = lane & 15;
  const int hi    = lane >> 4;       // 0: lanes 0-15, 1: lanes 16-31
  const int nbase = wv * 48;         // 16 waves * 48 cols = 768
  const int kr    = hi ? 16 : 0;     // B fragment K split: 0-15 / 16-31

  const unsigned short* wb0 = Wb + (size_t)(nbase + lm) * DN + kr;
  const unsigned short* wg0 = Wg + (size_t)(nbase + lm) * DN + kr;

  v8f accb[3] = {};
  v8f accg[3] = {};

  v16bf fbA[3], fgA[3], fbB[3], fgB[3];

  // Prologue: stage A <- kk = 0
  #pragma unroll
  for (int t = 0; t < 3; ++t) {
    fbA[t] = *reinterpret_cast<const v16bf*>(wb0 + (size_t)t * 16 * DN);
    fgA[t] = *reinterpret_cast<const v16bf*>(wg0 + (size_t)t * 16 * DN);
  }

  for (int kk = 0; kk < DN / 32; kk += 2) {
    // Stage B <- kk+1 (issued before consuming stage A)
    const int eb1 = (kk + 1) * 32;
    #pragma unroll
    for (int t = 0; t < 3; ++t) {
      fbB[t] = *reinterpret_cast<const v16bf*>(wb0 + (size_t)t * 16 * DN + eb1);
      fgB[t] = *reinterpret_cast<const v16bf*>(wg0 + (size_t)t * 16 * DN + eb1);
    }
    const v16bf afA =
        *reinterpret_cast<const v16bf*>(&A_lds[kk * 512 + lane * 16]);
    #pragma unroll
    for (int t = 0; t < 3; ++t) {
      accb[t] = __builtin_amdgcn_wmma_f32_16x16x32_bf16(
                    false, afA, false, fbA[t], (short)0, accb[t], false, false);
      accg[t] = __builtin_amdgcn_wmma_f32_16x16x32_bf16(
                    false, afA, false, fgA[t], (short)0, accg[t], false, false);
    }
    // Stage A <- kk+2 (clamped to a valid address on the tail; result unused)
    const int eb2 = (kk + 2 < DN / 32 * 1) ? 0 : 0; // placeholder, real below
    const int ebn = (kk + 2 < (DN / 32)) ? (kk + 2) * 32 : 0;
    (void)eb2;
    #pragma unroll
    for (int t = 0; t < 3; ++t) {
      fbA[t] = *reinterpret_cast<const v16bf*>(wb0 + (size_t)t * 16 * DN + ebn);
      fgA[t] = *reinterpret_cast<const v16bf*>(wg0 + (size_t)t * 16 * DN + ebn);
    }
    const v16bf afB =
        *reinterpret_cast<const v16bf*>(&A_lds[(kk + 1) * 512 + lane * 16]);
    #pragma unroll
    for (int t = 0; t < 3; ++t) {
      accb[t] = __builtin_amdgcn_wmma_f32_16x16x32_bf16(
                    false, afB, false, fbB[t], (short)0, accb[t], false, false);
      accg[t] = __builtin_amdgcn_wmma_f32_16x16x32_bf16(
                    false, afB, false, fgB[t], (short)0, accg[t], false, false);
    }
  }

  // Epilogue: out = (x - mean) * rstd * (gacc + gamma) + (bacc + beta)
  #pragma unroll
  for (int t = 0; t < 3; ++t) {
    const int d = nbase + t * 16 + lm;
    const float gam = wgamma[d];
    const float bet = wbeta[d];
    #pragma unroll
    for (int v = 0; v < 8; ++v) {
      const int  mr = v + hi * 8;    // C/D layout: VGPR v -> M=v (+8 for hi lanes)
      const long r  = r0 + mr;
      const float x = xin[r * DN + d];
      xout[r * DN + d] =
          (x - s_mean[mr]) * s_rstd[mr] * (accg[t][v] + gam) + (accb[t][v] + bet);
    }
  }
}

// ---------------------------------------------------------------------------
// 3) rel_guided = CLN(rel_hs | ent_ctx), ent_ctx built on the fly per tile
// ---------------------------------------------------------------------------
__global__ void __launch_bounds__(512)
k_cln_gemm_egr(const float* __restrict__ erowT, const float* __restrict__ ecolT,
               const unsigned short* __restrict__ Wb,
               const unsigned short* __restrict__ Wg,
               const float* __restrict__ wbeta, const float* __restrict__ wgamma,
               const float* __restrict__ xin, const float* __restrict__ mean_a,
               const float* __restrict__ rstd_a, float* __restrict__ xout)
{
  __shared__ __align__(64) unsigned short A_lds[24 * 512];
  __shared__ float s_mean[16], s_rstd[16];

  const int  tid = threadIdx.x;
  const long r0  = (long)blockIdx.x * 16;         // flattened (b,i,j) row base
  const int  b   = (int)(r0 / (MN * MN));
  const int  rem = (int)(r0 - (long)b * MN * MN);
  const int  i   = rem / MN;
  const int  jb  = rem - i * MN;                  // multiple of 16

  const float* er = erowT + (size_t)(b * MN + i) * EN;
  for (int idx = tid; idx < 16 * EN; idx += 512) {
    const int m = idx / EN;
    const int e = idx - m * EN;
    const float v = er[e] * ecolT[(size_t)(b * MN + jb + m) * EN + e];
    A_lds[a_lds_index(m, e)] = f2bf(v);
  }
  if (tid < 16) { s_mean[tid] = mean_a[r0 + tid]; s_rstd[tid] = rstd_a[r0 + tid]; }
  __syncthreads();

  cln_wave_core(A_lds, s_mean, s_rstd, Wb, Wg, wbeta, wgamma, xin, xout, r0);
}

// ---------------------------------------------------------------------------
// 4) rrowT[b,p,r] = sum_q RG[b,p,q,r]*alpha[r,q]
//    rcolT[b,p,r] = sum_q beta[r,q]*RG[b,q,p,r]
// ---------------------------------------------------------------------------
__global__ void __launch_bounds__(256)
k_bilin_rel(const float* __restrict__ RG, const float* __restrict__ alpha,
            const float* __restrict__ beta, float* __restrict__ rrowT,
            float* __restrict__ rcolT)
{
  const int bp = blockIdx.x;
  const int b  = bp >> 7;
  const int p  = bp & (MN - 1);
  for (int rc = 0; rc < DN / 256; ++rc) {
    const int r = rc * 256 + threadIdx.x;
    float ar = 0.f, ac = 0.f;
    for (int q = 0; q < MN; ++q) {
      const float h1 = RG[(((size_t)b * MN + p) * MN + q) * DN + r];
      const float h2 = RG[(((size_t)b * MN + q) * MN + p) * DN + r];
      ar += h1 * alpha[(size_t)r * MN + q];
      ac += h2 * beta[(size_t)r * MN + q];
    }
    rrowT[(size_t)bp * DN + r] = ar;
    rcolT[(size_t)bp * DN + r] = ac;
  }
}

// ---------------------------------------------------------------------------
// 5) ent_guided = CLN(ent_hs | lam-blended triangle rel_ctx)
// ---------------------------------------------------------------------------
__global__ void __launch_bounds__(512)
k_cln_gemm_rge(const float* __restrict__ rrowT, const float* __restrict__ rcolT,
               const float* __restrict__ lam, const int* __restrict__ upper,
               const unsigned short* __restrict__ Wb,
               const unsigned short* __restrict__ Wg,
               const float* __restrict__ wbeta, const float* __restrict__ wgamma,
               const float* __restrict__ xin, const float* __restrict__ mean_a,
               const float* __restrict__ rstd_a, float* __restrict__ xout)
{
  __shared__ __align__(64) unsigned short A_lds[24 * 512];
  __shared__ float s_mean[16], s_rstd[16];
  __shared__ int s_i[16], s_j[16];

  const int  tid = threadIdx.x;
  const long r0  = (long)blockIdx.x * 16;         // flattened (b,s) row base
  const int  b   = (int)(r0 / SN);
  const int  s0  = (int)(r0 - (long)b * SN);

  if (tid < 16) {
    const int u = upper[s0 + tid];
    const int ii = u / MN;
    s_i[tid] = ii;
    s_j[tid] = u - ii * MN;
    s_mean[tid] = mean_a[r0 + tid];
    s_rstd[tid] = rstd_a[r0 + tid];
  }
  __syncthreads();

  for (int idx = tid; idx < 16 * DN; idx += 512) {
    const int m = idx / DN;
    const int e = idx - m * DN;
    const int ii = s_i[m], jj = s_j[m];
    const float l  = lam[e];
    const float up = rrowT[(size_t)(b * MN + ii) * DN + e] *
                     rcolT[(size_t)(b * MN + jj) * DN + e];
    const float lo = rrowT[(size_t)(b * MN + jj) * DN + e] *
                     rcolT[(size_t)(b * MN + ii) * DN + e];
    A_lds[a_lds_index(m, e)] = f2bf(l * up + (1.f - l) * lo);
  }
  __syncthreads();

  cln_wave_core(A_lds, s_mean, s_rstd, Wb, Wg, wbeta, wgamma, xin, xout, r0);
}

// ---------------------------------------------------------------------------
extern "C" void kernel_launch(void* const* d_in, const int* in_sizes, int n_in,
                              void* d_out, int out_size, void* d_ws, size_t ws_size,
                              hipStream_t stream) {
  (void)in_sizes; (void)n_in; (void)out_size; (void)ws_size;

  const float* ent_hs    = (const float*)d_in[0];   // [B,S,E]
  const float* rel_hs    = (const float*)d_in[1];   // [B,M,M,R]
  const float* ent_alpha = (const float*)d_in[2];   // [E,M]
  const float* ent_beta  = (const float*)d_in[3];   // [E,M]
  const float* rel_alpha = (const float*)d_in[4];   // [R,M]
  const float* rel_beta  = (const float*)d_in[5];   // [R,M]
  const float* lam       = (const float*)d_in[6];   // [R]
  const float* egr_Wb    = (const float*)d_in[7];   // [R,E]
  const float* egr_Wg    = (const float*)d_in[8];
  const float* egr_beta  = (const float*)d_in[9];
  const float* egr_gamma = (const float*)d_in[10];
  const float* rge_Wb    = (const float*)d_in[11];  // [E,R]
  const float* rge_Wg    = (const float*)d_in[12];
  const float* rge_beta  = (const float*)d_in[13];
  const float* rge_gamma = (const float*)d_in[14];
  const int*   mirror    = (const int*)d_in[15];    // [M*M]
  const int*   upper     = (const int*)d_in[16];    // [S]
  // d_in[17] (lower_ids) derivable -> unused

  float* entg = (float*)d_out;                                  // [B,S,E]
  float* relg = (float*)d_out + (size_t)BN * SN * EN;           // [B,M,M,R]

  float* ws    = (float*)d_ws;
  float* erowT = ws;                                            // B*M*E
  float* ecolT = erowT + (size_t)BN * MN * EN;                  // B*M*E
  float* rmean = ecolT + (size_t)BN * MN * EN;                  // B*M*M
  float* rrstd = rmean + (size_t)BN * MN * MN;                  // B*M*M
  float* rrowT = rrstd + (size_t)BN * MN * MN;                  // B*M*R
  float* rcolT = rrowT + (size_t)BN * MN * DN;                  // B*M*R
  float* emean = rcolT + (size_t)BN * MN * DN;                  // B*S
  float* erstd = emean + (size_t)BN * SN;                       // B*S
  // bf16 weight copies (region starts 32B-aligned: float prefix is 7,079,936 B)
  unsigned short* egrWb_bf = (unsigned short*)(erstd + (size_t)BN * SN);
  unsigned short* egrWg_bf = egrWb_bf + NW;
  unsigned short* rgeWb_bf = egrWg_bf + NW;
  unsigned short* rgeWg_bf = rgeWb_bf + NW;

  // One-time weight conversions (NW/2 = 294912 packed pairs -> 1152 blocks)
  k_cvt_bf16<<<NW / 512, 256, 0, stream>>>(egr_Wb, (unsigned int*)egrWb_bf);
  k_cvt_bf16<<<NW / 512, 256, 0, stream>>>(egr_Wg, (unsigned int*)egrWg_bf);
  k_cvt_bf16<<<NW / 512, 256, 0, stream>>>(rge_Wb, (unsigned int*)rgeWb_bf);
  k_cvt_bf16<<<NW / 512, 256, 0, stream>>>(rge_Wg, (unsigned int*)rgeWg_bf);

  k_bilin_ent<<<BN * MN, 256, 0, stream>>>(ent_hs, ent_alpha, ent_beta, mirror,
                                           erowT, ecolT);
  k_row_stats<<<BN * MN * MN, 256, 0, stream>>>(rel_hs, rmean, rrstd);
  k_cln_gemm_egr<<<BN * MN * MN / 16, 512, 0, stream>>>(
      erowT, ecolT, egrWb_bf, egrWg_bf, egr_beta, egr_gamma, rel_hs, rmean,
      rrstd, relg);
  k_bilin_rel<<<BN * MN, 256, 0, stream>>>(relg, rel_alpha, rel_beta, rrowT,
                                           rcolT);
  k_row_stats<<<BN * SN, 256, 0, stream>>>(ent_hs, emean, erstd);
  k_cln_gemm_rge<<<BN * SN / 16, 512, 0, stream>>>(
      rrowT, rcolT, lam, upper, rgeWb_bf, rgeWg_bf, rge_beta, rge_gamma, ent_hs,
      emean, erstd, entg);
}
